// RoPEMaskedAttentionHead_1408749273570
// MI455X (gfx1250) — compile-verified, matrix operand order checked by reading, not run on hardware
//
#include <hip/hip_runtime.h>
#include <hip/hip_bf16.h>
#include <stdint.h>

// ---------------------------------------------------------------------------
// RoPE attention head for MI455X (gfx1250, wave32, WMMA).
// B=4, M=1024, D=512.  bf16 WMMA (f32 accum) for all matmuls; dense R is
// read sparsely (only cos/sin entries).  Flash-attention with online softmax;
// K chunks staged into LDS with gfx1250 async-to-LDS copies.
// ---------------------------------------------------------------------------

typedef __attribute__((ext_vector_type(8)))  float  v8f;
typedef __attribute__((ext_vector_type(8)))  __bf16 v8bf;
typedef __attribute__((ext_vector_type(16))) __bf16 v16bf;
typedef __attribute__((ext_vector_type(4)))  int    v4i;

#define B_  4
#define M_  1024
#define D_  512
#define N_  (B_ * M_)        // 4096 rows total
#define D2_ (D_ / 2)         // 256

// f32 -> bf16 round-to-nearest-even, as raw u16
__device__ __forceinline__ unsigned short f2bf_u(float f) {
    union { float f; uint32_t u; } v; v.f = f;
    uint32_t u = v.u;
    uint32_t r = u + 0x7FFFu + ((u >> 16) & 1u);
    return (unsigned short)(r >> 16);
}

// ---- gfx1250 async global->LDS copy (16 bytes per lane) -------------------
__device__ __forceinline__ void async_copy_b128(const void* gsrc, void* ldst) {
#if __has_builtin(__builtin_amdgcn_global_load_async_to_lds_b128)
    __builtin_amdgcn_global_load_async_to_lds_b128(
        (__attribute__((address_space(1))) v4i*)gsrc,
        (__attribute__((address_space(3))) v4i*)ldst, 0, 0);
#else
    unsigned lds_off = (unsigned)(uintptr_t)ldst;  // low 32 bits = LDS addr
    asm volatile("global_load_async_to_lds_b128 %0, %1, off"
                 :: "v"(lds_off), "v"(gsrc) : "memory");
#endif
}

__device__ __forceinline__ void wait_asynccnt0() {
#if __has_builtin(__builtin_amdgcn_s_wait_asynccnt)
    __builtin_amdgcn_s_wait_asynccnt(0);
#else
    asm volatile("s_wait_asynccnt 0x0" ::: "memory");
#endif
}

// Load one 16x32 bf16 operand tile (A-packing, ISA 7.12.2):
//   lane<16 : row M = lane,    K = {k0+0..7, k0+16..23}
//   lane>=16: row M = lane-16, K = {k0+8..15, k0+24..31}
// `base` points at row 0 of a row-major bf16 matrix with leading dim ldRow.
__device__ __forceinline__ v16bf load_ab(const unsigned short* base, int ldRow,
                                         int row_of_lane, int k0, int lane) {
    int ko = k0 + ((lane & 16) ? 8 : 0);
    const unsigned short* rp = base + (size_t)row_of_lane * ldRow + ko;
    v8bf c0 = *(const v8bf*)(rp);
    v8bf c1 = *(const v8bf*)(rp + 16);
    v16bf a;
#pragma unroll
    for (int i = 0; i < 8; ++i) { a[i] = c0[i]; a[8 + i] = c1[i]; }
    return a;
}

// ---------------------------------------------------------------------------
// Kernel 0: bf16 conversions + sparse cos/sin extraction from dense R.
//   cos[m,i] = R[m, 2i,   2i]
//   sin[m,i] = R[m, 2i+1, 2i]
// ---------------------------------------------------------------------------
__global__ __launch_bounds__(256) void prep_kernel(
    const float* __restrict__ x,  const float* __restrict__ wq,
    const float* __restrict__ wk, const float* __restrict__ wv,
    const float* __restrict__ R,
    unsigned short* __restrict__ XBF, unsigned short* __restrict__ WBF,
    float* __restrict__ COSA, float* __restrict__ SINA) {
    int i = blockIdx.x * 256 + threadIdx.x;
    const int NX = N_ * D_;      // 2097152
    const int NW = D_ * D_;      // 262144
    if (i < NX) { XBF[i] = f2bf_u(x[i]); return; }
    i -= NX;
    if (i < 3 * NW) {
        int mat = i / NW, k = i % NW;
        const float* w = (mat == 0) ? wq : ((mat == 1) ? wk : wv);
        WBF[i] = f2bf_u(w[k]);
        return;
    }
    i -= 3 * NW;
    if (i < M_ * D2_) {
        int m  = i >> 8;         // / 256
        int ii = i & 255;
        size_t base = ((size_t)m * D_ + 2 * ii) * D_ + 2 * ii;
        COSA[i] = R[base];        // R[m, 2i,   2i]  =  cos
        SINA[i] = R[base + D_];   // R[m, 2i+1, 2i]  =  sin
    }
}

// ---------------------------------------------------------------------------
// Kernel 1: QKV projection (NT GEMM, bf16 WMMA) + RoPE, writes:
//   QROT, KROT : [B*M, D] bf16 (rotated)
//   VT         : [B, D, M] bf16 (transposed V, for PV matmul B-operand)
// One wave per 16x16 C tile; 3*256*32 = 24576 tiles total.
// ---------------------------------------------------------------------------
__global__ __launch_bounds__(256) void qkv_rope_kernel(
    const unsigned short* __restrict__ XBF,
    const unsigned short* __restrict__ WBF,
    const float* __restrict__ COSA, const float* __restrict__ SINA,
    unsigned short* __restrict__ QROT, unsigned short* __restrict__ KROT,
    unsigned short* __restrict__ VT) {
    int lane = threadIdx.x & 31;
    int gw   = blockIdx.x * 8 + (threadIdx.x >> 5);
    int mat  = gw / 8192;            // 0=Q 1=K 2=V
    int t    = gw % 8192;
    int row0 = (t >> 5) * 16;        // output row block (n dimension)
    int col0 = (t & 31) * 16;        // output col block (e dimension)

    const unsigned short* Wm = WBF + (size_t)mat * (D_ * D_);
    int arow = row0 + (lane & 15);
    int brow = col0 + (lane & 15);

    v8f acc = {};
#pragma unroll 4
    for (int dc = 0; dc < 16; ++dc) {
        v16bf a = load_ab(XBF, D_, arow, dc * 32, lane);
        v16bf b = load_ab(Wm,  D_, brow, dc * 32, lane);
        acc = __builtin_amdgcn_wmma_f32_16x16x32_bf16(
            false, a, false, b, (short)0, acc, false, false);
    }

    if (mat == 2) {
        // store V transposed: VT[b, e, m]
        int e = col0 + (lane & 15);
#pragma unroll
        for (int r = 0; r < 8; ++r) {
            int n  = row0 + r + ((lane & 16) ? 8 : 0);
            int bb = n >> 10, m = n & 1023;
            VT[((size_t)(bb * D_ + e)) * M_ + m] = f2bf_u(acc[r]);
        }
    } else {
        unsigned short* OUT = (mat == 0) ? QROT : KROT;
        int j = col0 + (lane & 15);
        int jh = j >> 1;
        bool even = (j & 1) == 0;
#pragma unroll
        for (int r = 0; r < 8; ++r) {
            int n = row0 + r + ((lane & 16) ? 8 : 0);
            int m = n & 1023;
            float c = COSA[m * D2_ + jh];
            float s = SINA[m * D2_ + jh];
            float v = acc[r];
            float p = __shfl_xor(v, 1, 32);  // even<->odd column partner
            // q_rot[2i]   =  q[2i]*cos + q[2i+1]*sin
            // q_rot[2i+1] = -q[2i]*sin + q[2i+1]*cos
            float o = even ? (v * c + p * s) : (v * c - p * s);
            OUT[(size_t)n * D_ + j] = f2bf_u(o);
        }
    }
}

// ---------------------------------------------------------------------------
// Kernel 2: flash attention over M=1024 keys, 32 q-rows per workgroup.
// 8 waves: waves 0..3 stage K chunk into LDS via async copies, then compute
//          S = Q.K^T (scaled) into LDS;
//          waves 4..7 do online softmax + P.V accumulation (16q x 256e each).
// Barriers per chunk: bar0 (K staged), bar1 (S ready), bar2 (P ready).
// ---------------------------------------------------------------------------
__global__ __launch_bounds__(256) void attn_kernel(
    const unsigned short* __restrict__ QROT,
    const unsigned short* __restrict__ KROT,
    const unsigned short* __restrict__ VT,
    float* __restrict__ OUT) {
    __shared__ __align__(16) unsigned short sK[32 * D_];   // 32 KB staged K chunk
    __shared__ __align__(16) float          sS[32 * 32];
    __shared__ __align__(16) unsigned short sP[32 * 32];
    __shared__ float sAlpha[32];
    __shared__ float sL[32];

    int lane = threadIdx.x & 31;
    int wave = threadIdx.x >> 5;
    int blk  = blockIdx.x;
    int b    = blk >> 5;
    int q0   = (blk & 31) * 32;

    const unsigned short* Q = QROT + ((size_t)(b * M_ + q0)) * D_;
    const unsigned short* K = KROT + (size_t)b * M_ * D_;
    const unsigned short* V = VT   + (size_t)b * D_ * M_;
    float* Ob = OUT + ((size_t)(b * M_ + q0)) * D_;
    const float scale = 0.04419417382415922f;  // 1/sqrt(512)

    if (wave < 4) {
        // ---- S waves: async-stage K, one 16x16 tile of S per key-chunk ----
        int qsub = wave & 1, ksub = wave >> 1;
        int arow = qsub * 16 + (lane & 15);
        v16bf qreg[16];
#pragma unroll
        for (int dc = 0; dc < 16; ++dc)
            qreg[dc] = load_ab(Q, D_, arow, dc * 32, lane);

        for (int kc = 0; kc < 32; ++kc) {
            // stage 32 contiguous K rows (32 KB) into sK; this wave's quarter
            // is 8 KB = 16 async b128 instructions (512 B each).
            {
                const char* gbase = (const char*)(K + (size_t)(kc * 32) * D_);
                char* lbase = (char*)sK;
#pragma unroll
                for (int i = 0; i < 16; ++i) {
                    size_t boff = (size_t)wave * 8192 + (size_t)i * 512 + (size_t)lane * 16;
                    async_copy_b128(gbase + boff, lbase + boff);
                }
            }
            wait_asynccnt0();
            __syncthreads();   // bar0: sK fully staged

            int brow = ksub * 16 + (lane & 15);   // row within staged chunk
            v8f s = {};
#pragma unroll 4
            for (int dc = 0; dc < 16; ++dc) {
                v16bf kb = load_ab(sK, D_, brow, dc * 32, lane);  // ds reads
                s = __builtin_amdgcn_wmma_f32_16x16x32_bf16(
                    false, qreg[dc], false, kb, (short)0, s, false, false);
            }
            int coln = ksub * 16 + (lane & 15);
#pragma unroll
            for (int r = 0; r < 8; ++r) {
                int row = qsub * 16 + r + ((lane & 16) ? 8 : 0);
                sS[row * 32 + coln] = s[r] * scale;
            }
            __syncthreads();   // bar1: S visible
            __syncthreads();   // bar2: PV done reading S / P ready
        }
        __syncthreads();       // final
    } else {
        // ---- PV waves: online softmax + P.V ----
        int v    = wave - 4;
        int qsub = v & 1;
        int e0   = (v >> 1) * 256;
        int myrow = qsub * 16 + (lane & 15);

        v8f acc[16];
#pragma unroll
        for (int t = 0; t < 16; ++t) acc[t] = (v8f){};
        float m_run = -3.0e38f, l_run = 0.f;

        for (int kc = 0; kc < 32; ++kc) {
            __syncthreads();   // bar0: K staged (no-op for PV)
            __syncthreads();   // bar1: sS ready
            float mx = m_run;
            float sv[32];
#pragma unroll
            for (int j = 0; j < 32; ++j) {
                sv[j] = sS[myrow * 32 + j];
                mx = fmaxf(mx, sv[j]);
            }
            float alpha = __expf(m_run - mx);
            float sum = 0.f;
#pragma unroll
            for (int j = 0; j < 32; ++j) {
                float p = __expf(sv[j] - mx);
                sum += p;
                sP[myrow * 32 + j] = f2bf_u(p);
            }
            l_run = l_run * alpha + sum;
            m_run = mx;
            sAlpha[myrow] = alpha;
            __syncthreads();   // bar2: sP/sAlpha ready, release sS

            float af[8];
#pragma unroll
            for (int r = 0; r < 8; ++r)
                af[r] = sAlpha[qsub * 16 + r + ((lane & 16) ? 8 : 0)];
#pragma unroll
            for (int t = 0; t < 16; ++t)
#pragma unroll
                for (int r = 0; r < 8; ++r) acc[t][r] *= af[r];

            v16bf pa = load_ab(sP, 32, myrow, 0, lane);  // A = P (16x32)
            int k0 = kc * 32;
            if (kc < 31)
                __builtin_prefetch(V + (size_t)(e0 + (lane & 15)) * M_ + k0 + 32, 0, 1);
#pragma unroll
            for (int t = 0; t < 16; ++t) {
                int erow = e0 + t * 16 + (lane & 15);
                v16bf vb = load_ab(V, M_, erow, k0, lane);  // B = V tile
                acc[t] = __builtin_amdgcn_wmma_f32_16x16x32_bf16(
                    false, pa, false, vb, (short)0, acc[t], false, false);
            }
        }
        sL[myrow] = l_run;
        __syncthreads();       // final
        float lf[8];
#pragma unroll
        for (int r = 0; r < 8; ++r)
            lf[r] = 1.f / sL[qsub * 16 + r + ((lane & 16) ? 8 : 0)];
#pragma unroll
        for (int t = 0; t < 16; ++t) {
            int ecol = e0 + t * 16 + (lane & 15);
#pragma unroll
            for (int r = 0; r < 8; ++r) {
                int row = qsub * 16 + r + ((lane & 16) ? 8 : 0);
                Ob[(size_t)row * D_ + ecol] = acc[t][r] * lf[r];
            }
        }
    }
}

// ---------------------------------------------------------------------------
extern "C" void kernel_launch(void* const* d_in, const int* in_sizes, int n_in,
                              void* d_out, int out_size, void* d_ws, size_t ws_size,
                              hipStream_t stream) {
    (void)in_sizes; (void)n_in; (void)out_size; (void)ws_size;
    const float* x  = (const float*)d_in[0];
    const float* wq = (const float*)d_in[1];
    const float* wk = (const float*)d_in[2];
    const float* wv = (const float*)d_in[3];
    const float* R  = (const float*)d_in[4];
    float* out = (float*)d_out;

    char* ws = (char*)d_ws;
    size_t off = 0;
    auto alloc = [&](size_t bytes) {
        size_t o = off;
        off += (bytes + 255) & ~(size_t)255;
        return o;
    };
    unsigned short* XBF  = (unsigned short*)(ws + alloc((size_t)N_ * D_ * 2));
    unsigned short* WBF  = (unsigned short*)(ws + alloc((size_t)3 * D_ * D_ * 2));
    float*          COSA = (float*)(ws + alloc((size_t)M_ * D2_ * 4));
    float*          SINA = (float*)(ws + alloc((size_t)M_ * D2_ * 4));
    unsigned short* QROT = (unsigned short*)(ws + alloc((size_t)N_ * D_ * 2));
    unsigned short* KROT = (unsigned short*)(ws + alloc((size_t)N_ * D_ * 2));
    unsigned short* VT   = (unsigned short*)(ws + alloc((size_t)B_ * D_ * M_ * 2));

    const int prepN = N_ * D_ + 3 * D_ * D_ + M_ * D2_;  // 3,145,728
    prep_kernel<<<(prepN + 255) / 256, 256, 0, stream>>>(
        x, wq, wk, wv, R, XBF, WBF, COSA, SINA);

    // 3 matrices * 8192 tiles = 24576 waves = 3072 blocks of 8 waves
    qkv_rope_kernel<<<3072, 256, 0, stream>>>(
        XBF, WBF, COSA, SINA, QROT, KROT, VT);

    // B * (M/32) = 128 workgroups
    attn_kernel<<<128, 256, 0, stream>>>(QROT, KROT, VT, out);
}
